// ESN_23012434772618
// MI455X (gfx1250) — compile-verified, matrix operand order checked by reading, not run on hardware
//
#include <hip/hip_runtime.h>
#include <hip/hip_bf16.h>
#include <math.h>

// ---- problem constants (from reference) ----
#define NB    128      // batch
#define NN    1024     // reservoir size
#define NI    64       // input size
#define NT    512      // time steps
#define ALPHA 0.5f

typedef __attribute__((ext_vector_type(16))) __bf16 v16bf;
typedef __attribute__((ext_vector_type(8)))  float  v8f;
typedef __attribute__((ext_vector_type(4)))  int    v4i;
typedef __attribute__((ext_vector_type(4)))  unsigned int u32x4;

typedef __attribute__((address_space(1))) v4i* gptr_b128;   // global int4*
typedef __attribute__((address_space(3))) v4i* lptr_b128;   // LDS int4*

__device__ __forceinline__ v8f wmma_bf16(v16bf a, v16bf b, v8f c) {
    // D = A(16x32 bf16) * B(32x16 bf16) + C(16x16 f32)
    return __builtin_amdgcn_wmma_f32_16x16x32_bf16(
        /*neg_a=*/false, a, /*neg_b=*/false, b,
        /*c_mod=*/(short)0, c, /*reuse_a=*/false, /*reuse_b=*/false);
}

// ---- CDNA5 async global->LDS copy (ASYNCcnt-tracked), with sync fallback ----
#if __has_builtin(__builtin_amdgcn_global_load_async_to_lds_b128)
#define ESN_ASYNC 1
#else
#define ESN_ASYNC 0
#endif

__device__ __forceinline__ void async_copy_b128(const __bf16* g, __bf16* l) {
#if ESN_ASYNC
    __builtin_amdgcn_global_load_async_to_lds_b128(
        (gptr_b128)(v4i*)(__bf16*)g,
        (lptr_b128)(v4i*)l,
        /*imm offset=*/0, /*cpol=*/0);
#else
    *(u32x4*)l = *(const u32x4*)g;   // synchronous fallback through VGPRs
#endif
}

template <int N>
__device__ __forceinline__ void wait_async() {
#if ESN_ASYNC
 #if __has_builtin(__builtin_amdgcn_s_wait_asynccnt)
    __builtin_amdgcn_s_wait_asynccnt(N);
 #else
    asm volatile("s_wait_asynccnt %0" :: "i"(N) : "memory");
 #endif
#endif
}

// Fragment packing convention (wave32, 16-bit A/B operands, K-block = 32):
//   frag = 1KB = 512 halves; per lane l: 16 halves at [l*32B .. l*32B+32B)
//   half h within lane l maps to k' = (h&7) + ((h&8)?16:0) + ((l&16)?8:0)
//   A-frag:  lane row m = l&15          (16x32 tile, k = kb*32 + k')
//   B-frag:  lane col n = base + (l&15) (32x16 tile, k = kb*32 + k')

// ---- prep: pack W [k][n] f32 -> B-fragment-major bf16 ----
__global__ void esn_pack_W(const float* __restrict__ W, __bf16* __restrict__ dst) {
    unsigned idx = blockIdx.x * blockDim.x + threadIdx.x;
    if (idx >= (unsigned)(64 * 32 * 512)) return;
    unsigned h = idx & 15, l = (idx >> 4) & 31, f = idx >> 9;
    unsigned nt = f >> 5, kb = f & 31;
    unsigned n  = nt * 16 + (l & 15);
    unsigned kp = (h & 7) + ((h & 8) ? 16u : 0u) + ((l & 16) ? 8u : 0u);
    unsigned k  = kb * 32 + kp;
    dst[idx] = (__bf16)W[(size_t)k * NN + n];
}

// ---- prep: pack W_in [i][n] f32 -> B-fragment-major bf16 (K=64 -> kb 0..1) ----
__global__ void esn_pack_Win(const float* __restrict__ Win, __bf16* __restrict__ dst) {
    unsigned idx = blockIdx.x * blockDim.x + threadIdx.x;
    if (idx >= (unsigned)(64 * 2 * 512)) return;
    unsigned h = idx & 15, l = (idx >> 4) & 31, f = idx >> 9;
    unsigned nt = f >> 1, kb = f & 1;
    unsigned n  = nt * 16 + (l & 15);
    unsigned kp = (h & 7) + ((h & 8) ? 16u : 0u) + ((l & 16) ? 8u : 0u);
    unsigned i  = kb * 32 + kp;
    dst[idx] = (__bf16)Win[(size_t)i * NN + n];
}

// ---- prep: pack Input [b][i][t] f32 -> per-(t, batch-tile) A-fragment bf16 ----
__global__ void esn_pack_In(const float* __restrict__ In, __bf16* __restrict__ dst) {
    unsigned idx = blockIdx.x * blockDim.x + threadIdx.x;
    if (idx >= (unsigned)(NT * 8 * 2 * 512)) return;
    unsigned h = idx & 15, l = (idx >> 4) & 31, f = idx >> 9;
    unsigned kb2 = f & 1, bt = (f >> 1) & 7, t = f >> 4;
    unsigned m  = l & 15;
    unsigned kp = (h & 7) + ((h & 8) ? 16u : 0u) + ((l & 16) ? 8u : 0u);
    unsigned i  = kb2 * 32 + kp;
    unsigned b  = bt * 16 + m;
    dst[idx] = (__bf16)In[((size_t)b * NI + i) * NT + t];
}

// ---- main persistent ESN kernel ----
// grid = 8 workgroups (one per 16-row batch tile), block = 512 threads = 16 waves.
// Each wave owns 4 column tiles. State x: f32 in accumulator-layout registers +
// bf16 A-fragment copy double-buffered in LDS.
// W stream pipeline:
//   LDS:  4-slot rotating buffer, async global->LDS staged 3 k-blocks ahead
//   regs: A/B fragments for kb+1 ds-loaded before the WMMA group of kb
__global__ __launch_bounds__(512) void esn_main(
    const __bf16* __restrict__ packedW,
    const __bf16* __restrict__ packedWin,
    const __bf16* __restrict__ packedIn,
    float* __restrict__ out)
{
    __shared__ __align__(32) __bf16 xbuf[2][32 * 512];          // 64 KB A fragments
    __shared__ __align__(32) __bf16 sB[16 * 4 * 4 * 512];       // 256 KB staged W frags

    const int bt   = blockIdx.x;
    const int wave = threadIdx.x >> 5;
    const int lane = threadIdx.x & 31;
    const int lm   = lane & 15;
    const int lhi  = lane >> 4;

    // zero initial state buffer (x0 = 0)
    for (int i = threadIdx.x; i < 32 * 512; i += 512)
        xbuf[0][i] = (__bf16)0.0f;
    __syncthreads();

    // per-wave base of W fragment stream: frags (wave*4+j)*32 + kb
    const __bf16* wbase  = packedW + ((size_t)wave * 4 * 32) * 512 + lane * 16;
    const __bf16* winbse = packedWin + ((size_t)wave * 4 * 2) * 512 + lane * 16;
    __bf16*       sbase  = &sB[wave * 4 * 4 * 512] + lane * 16;

    // stage one k-block (4 B-frags, 8 async b128 per wave) into LDS slot `buf`
    auto stage = [&](int kb, int buf) {
#pragma unroll
        for (int j = 0; j < 4; ++j) {
            const __bf16* g = wbase + ((size_t)(j * 32 + kb)) * 512;
            __bf16*       l = sbase + (buf * 4 + j) * 512;
            async_copy_b128(g, l);
            async_copy_b128(g + 8, l + 8);
        }
    };

    v8f xold[4] = {};   // persistent f32 state, C/D layout, 4 tiles

    for (int t = 0; t < NT; ++t) {
        const int curb = t & 1, nxtb = curb ^ 1;

        // ---- async prologue: stage kb = 0,1,2 into LDS slots 0,1,2 ----
        stage(0, 0);
        stage(1, 1);
        stage(2, 2);

        // ---- input projection: s_t @ W_in (K=64), overlaps async latency ----
        v8f acc[4] = {};
        const __bf16* inFrag = packedIn + ((size_t)(t * 16 + bt * 2)) * 512;
#pragma unroll
        for (int kb2 = 0; kb2 < 2; ++kb2) {
            v16bf a = *(const v16bf*)(inFrag + kb2 * 512 + lane * 16);
#pragma unroll
            for (int j = 0; j < 4; ++j) {
                v16bf b = *(const v16bf*)(winbse + ((size_t)(j * 2 + kb2)) * 512);
                acc[j] = wmma_bf16(a, b, acc[j]);
            }
        }

        // ---- preload kb=0 fragments into registers ----
        v16bf areg[2];
        v16bf breg[2][4];
        wait_async<16>();                       // slot 0 landed (1,2 in flight)
        areg[0] = *(const v16bf*)(&xbuf[curb][lane * 16]);
#pragma unroll
        for (int j = 0; j < 4; ++j)
            breg[0][j] = *(const v16bf*)(sbase + j * 512);

        // ---- recurrent projection: x @ W (K=1024 -> 32 k-blocks) ----
#pragma unroll 4
        for (int kb = 0; kb < 32; ++kb) {
            const int cb = kb & 1, nb = cb ^ 1;
            // fetch kb+1 fragments into registers (their ds latency hides
            // behind the WMMA group below)
            if (kb < 31) {
                if (kb < 30) wait_async<8>(); else wait_async<0>();
                const int slot = (kb + 1) & 3;
                areg[nb] = *(const v16bf*)(&xbuf[curb][(kb + 1) * 512 + lane * 16]);
#pragma unroll
                for (int j = 0; j < 4; ++j)
                    breg[nb][j] = *(const v16bf*)(sbase + (slot * 4 + j) * 512);
            }
            // consume kb from registers
#pragma unroll
            for (int j = 0; j < 4; ++j)
                acc[j] = wmma_bf16(areg[cb], breg[cb][j], acc[j]);
            // stage kb+3 into the slot kb-1 vacated (its regs consumed above)
            if (kb < 29) stage(kb + 3, (kb + 3) & 3);
        }

        // ---- leaky tanh update, emit outputs, stage next-step A fragments ----
#pragma unroll
        for (int j = 0; j < 4; ++j) {
            const int n   = wave * 64 + j * 16 + lm;
            const int kpp = n & 31;
            const int kb  = n >> 5;
            const int hh  = (kpp & 7) + ((kpp & 16) ? 8 : 0);
            const int lp  = (kpp & 8) ? 16 : 0;
#pragma unroll
            for (int r = 0; r < 8; ++r) {
                const int m = r + 8 * lhi;
                float xn = (1.0f - ALPHA) * xold[j][r] + ALPHA * tanhf(acc[j][r]);
                xold[j][r] = xn;
                __builtin_nontemporal_store(
                    xn, out + (((size_t)(bt * 16 + m)) * NN + n) * NT + t);
                xbuf[nxtb][kb * 512 + (m + lp) * 16 + hh] = (__bf16)xn;
            }
        }
        __syncthreads();   // next-step fragments visible to all 16 waves
    }
}

extern "C" void kernel_launch(void* const* d_in, const int* in_sizes, int n_in,
                              void* d_out, int out_size, void* d_ws, size_t ws_size,
                              hipStream_t stream) {
    const float* Input = (const float*)d_in[0];   // [128, 64, 512] f32
    const float* W     = (const float*)d_in[1];   // [1024, 1024]   f32
    const float* Win   = (const float*)d_in[2];   // [64, 1024]     f32
    float*       out   = (float*)d_out;           // [128, 1024, 512] f32

    // workspace layout (halves): packedW 1M | packedWin 64K | packedIn 4M (~10.6 MB)
    __bf16* pW   = (__bf16*)d_ws;
    __bf16* pWin = pW + (size_t)(64 * 32 * 512);
    __bf16* pIn  = pWin + (size_t)(64 * 2 * 512);

    esn_pack_W  <<<(64 * 32 * 512) / 256, 256, 0, stream>>>(W, pW);
    esn_pack_Win<<<(64 * 2 * 512) / 256, 256, 0, stream>>>(Win, pWin);
    esn_pack_In <<<(NT * 8 * 2 * 512) / 256, 256, 0, stream>>>(Input, pIn);

    esn_main<<<8, 512, 0, stream>>>(pW, pWin, pIn, out);
}
